// CausalSelfAttention_63685775065097
// MI455X (gfx1250) — matrix operand order repaired and checked by reading, not compiled, not run on hardware
//
#include <hip/hip_runtime.h>
#include <stdint.h>

#define N_EMBD 1024
#define NH     16
#define HD     64
#define BB     4
#define TT     2048
#define MM     (BB*TT)   // 8192 rows

typedef __attribute__((ext_vector_type(16))) __bf16    v16bf;
typedef __attribute__((ext_vector_type(8)))  float     v8f;
typedef __attribute__((ext_vector_type(8)))  uint32_t  v8u;
typedef __attribute__((ext_vector_type(4)))  uint32_t  v4u;
typedef __attribute__((ext_vector_type(4)))  int       v4i;
typedef __attribute__((ext_vector_type(8)))  int       v8i;

__device__ __forceinline__ uint16_t f2bf(float f) {
  uint32_t u = __float_as_uint(f);
  u += 0x7FFFu + ((u >> 16) & 1u);          // round-to-nearest-even
  return (uint16_t)(u >> 16);
}
__device__ __forceinline__ uint32_t pack2bf(float a, float b) {
  return (uint32_t)f2bf(a) | ((uint32_t)f2bf(b) << 16);
}

__device__ __forceinline__ v8f wmma_bf16(v16bf a, v16bf b, v8f c) {
  // D = A(16x32) * B(32x16) + C, f32 accumulate
  return __builtin_amdgcn_wmma_f32_16x16x32_bf16(false, a, false, b, (short)0, c,
                                                 false, false);
}

// A-fragment (16x32 bf16). base = &row[ half*16 ]; 16 contiguous elems.
__device__ __forceinline__ v16bf ldfragA(const uint16_t* base) {
  uint4 x0 = *(const uint4*)(base);
  uint4 x1 = *(const uint4*)(base + 8);
  v8u f;
  f[0]=x0.x; f[1]=x0.y; f[2]=x0.z; f[3]=x0.w;
  f[4]=x1.x; f[5]=x1.y; f[6]=x1.z; f[7]=x1.w;
  return __builtin_bit_cast(v16bf, f);
}
// B-fragment (32x16 bf16). base = &col_row[ half*16 ]; 16 contiguous elems.
__device__ __forceinline__ v16bf ldfragB(const uint16_t* base) {
  uint4 x0 = *(const uint4*)(base);
  uint4 x1 = *(const uint4*)(base + 8);
  v8u f;
  f[0]=x0.x; f[1]=x0.y; f[2]=x0.z; f[3]=x0.w;
  f[4]=x1.x; f[5]=x1.y; f[6]=x1.z; f[7]=x1.w;
  return __builtin_bit_cast(v16bf, f);
}

__device__ __forceinline__ v8f zero8() {
  v8f z = {0.f,0.f,0.f,0.f,0.f,0.f,0.f,0.f};
  return z;
}

// ---- Tensor Data Mover: 2D tile (bf16) global -> LDS, with LDS row padding ----
// D# layout per CDNA5 ISA ch.8: group0 = {count/lds/global/type}, group1 = dims.
// pad_interval=3 -> pad after every 16 dwords (64B row); pad_amount=11 -> 12 dwords
// (48B), producing the 112B LDS row stride used by the fragment loaders.
__device__ __forceinline__ void tdm_load_tile_bf16(uint32_t lds_byte_off,
                                                   const void* gptr,
                                                   uint32_t tile_w,      // elems
                                                   uint32_t tile_h,      // rows
                                                   uint64_t row_stride)  // elems
{
  uint64_t ga = (uint64_t)(uintptr_t)gptr;
  v4u g0;
  g0.x = 1u;                                   // count=1 (valid user descriptor)
  g0.y = lds_byte_off;                         // lds_addr
  g0.z = (uint32_t)ga;                         // global_addr[31:0]
  g0.w = (uint32_t)(ga >> 32) | (2u << 30);    // global_addr[56:32] | type=2
  uint32_t w[8];
  uint32_t dim0 = (uint32_t)row_stride;        // tensor_dim0 (no clipping)
  uint32_t dim1 = tile_h;                      // tensor_dim1
  w[0] = (1u << 16)                            // data_size = 1 (2 bytes)
       | (1u << 20)                            // pad_enable
       | (3u << 22)                            // pad_interval: every 16 dwords
       | (11u << 25);                          // pad_amount: 12 dwords
  w[1] = (dim0 & 0xFFFFu) << 16;               // tensor_dim0[15:0] @ bits 63:48
  w[2] = (dim0 >> 16) | ((dim1 & 0xFFFFu) << 16);
  w[3] = (dim1 >> 16) | (tile_w << 16);        // tile_dim0 @ bits 127:112
  w[4] = tile_h & 0xFFFFu;                     // tile_dim1 (tile_dim2 = 0)
  w[5] = (uint32_t)row_stride;                 // tensor_dim0_stride[31:0]
  w[6] = (uint32_t)(row_stride >> 32) & 0xFFFFu;
  w[7] = 0;
  v8i g1;
  for (int i = 0; i < 8; i++) g1[i] = (int)w[i];
  v4i z4 = {0, 0, 0, 0};
#if defined(__clang_major__) && (__clang_major__ >= 23)
  v8i z8 = {0, 0, 0, 0, 0, 0, 0, 0};
  __builtin_amdgcn_tensor_load_to_lds(g0, g1, z4, z4, z8, 0);
#else
  __builtin_amdgcn_tensor_load_to_lds(g0, g1, z4, z4, 0);
#endif
}

// ---------------- prep kernels ----------------

__global__ void cvt_f32_bf16(const float* __restrict__ in,
                             uint16_t* __restrict__ out, int n) {
  int i = (blockIdx.x * blockDim.x + threadIdx.x) * 4;
  if (i < n) {
    float4 v = *(const float4*)(in + i);
    uint2 o;
    o.x = pack2bf(v.x, v.y);
    o.y = pack2bf(v.z, v.w);
    *(uint2*)(out + i) = o;
  }
}

// in: f32 [R][Cc] row-major  ->  out: bf16 [Cc][R]
__global__ __launch_bounds__(256)
void transpose_cvt(const float* __restrict__ in, uint16_t* __restrict__ out,
                   int R, int Cc) {
  __shared__ float tile[32][33];
  int tx = threadIdx.x & 31, ty = threadIdx.x >> 5;
  int r0 = blockIdx.y * 32, c0 = blockIdx.x * 32;
  for (int i = 0; i < 4; i++)
    tile[ty + i*8][tx] = in[(size_t)(r0 + ty + i*8) * Cc + c0 + tx];
  __syncthreads();
  for (int i = 0; i < 4; i++)
    out[(size_t)(c0 + ty + i*8) * R + r0 + tx] = f2bf(tile[tx][ty + i*8]);
}

// ---------------- generic bf16 WMMA GEMM (TDM-staged, double-buffered) ----------------
// C[m][n] = sum_k A[m][k]*Bt[n][k] + bias[n]
// MODE 0: scatter bf16 into q/k/v per-head layouts (V transposed [bh][d][t])
// MODE 1: f32 row-major to outF
template<int MODE>
__global__ __launch_bounds__(256)
void gemm_bf16_wmma(const uint16_t* __restrict__ A,
                    const uint16_t* __restrict__ Bt,
                    const float* __restrict__ bias,
                    float* __restrict__ outF,
                    uint16_t* __restrict__ q_ws,
                    uint16_t* __restrict__ k_ws,
                    uint16_t* __restrict__ v_ws,
                    int Nd, int Kd) {
  // 112-byte LDS row stride: conflict-free b128 fragment reads; two buffers.
  __shared__ uint16_t As[2][128 * 56];
  __shared__ uint16_t Bs[2][128 * 56];

  const int tid  = threadIdx.x;
  const int lane = tid & 31;
  const int wave = tid >> 5;
  const int wv   = __builtin_amdgcn_readfirstlane(wave);
  const int wm   = wv & 3;        // 4 waves along M (32 rows each)
  const int wn   = wv >> 2;       // 2 waves along N (64 cols each)
  const int hf   = lane >> 4;
  const int l16  = lane & 15;
  const int m0   = blockIdx.y * 128;
  const int n0   = blockIdx.x * 128;

  v8f acc[2][4];
  for (int i = 0; i < 2; i++)
    for (int j = 0; j < 4; j++) acc[i][j] = zero8();

  const int nk = Kd >> 5;
  if (wv == 0) {   // TDM issue wave (TDM ignores EXEC; descriptors are uniform)
    tdm_load_tile_bf16((uint32_t)(uintptr_t)&As[0][0],
                       A + (size_t)m0 * Kd, 32u, 128u, (uint64_t)Kd);
    tdm_load_tile_bf16((uint32_t)(uintptr_t)&Bs[0][0],
                       Bt + (size_t)n0 * Kd, 32u, 128u, (uint64_t)Kd);
  }

  for (int it = 0; it < nk; ++it) {
    const int buf = it & 1;
    __syncthreads();                 // everyone done reading buf^1 (prev iter)
    if (wv == 0) {
      if (it + 1 < nk) {             // prefetch next tile into the other buffer
        tdm_load_tile_bf16((uint32_t)(uintptr_t)&As[buf ^ 1][0],
                           A + (size_t)m0 * Kd + (it + 1) * 32, 32u, 128u,
                           (uint64_t)Kd);
        tdm_load_tile_bf16((uint32_t)(uintptr_t)&Bs[buf ^ 1][0],
                           Bt + (size_t)n0 * Kd + (it + 1) * 32, 32u, 128u,
                           (uint64_t)Kd);
        __builtin_amdgcn_s_wait_tensorcnt(2);  // current tile done; next in flight
      } else {
        __builtin_amdgcn_s_wait_tensorcnt(0);
      }
    }
    __syncthreads();                 // publish DMA'd tile to all waves

    v16bf af[2];
    for (int i = 0; i < 2; i++) {
      int m = wm * 32 + i * 16 + l16;
      af[i] = ldfragA(&As[buf][0] + m * 56 + hf * 16);
    }
    for (int j = 0; j < 4; j++) {
      int n = wn * 64 + j * 16 + l16;
      v16bf bfr = ldfragB(&Bs[buf][0] + n * 56 + hf * 16);
      for (int i = 0; i < 2; i++)
        acc[i][j] = wmma_bf16(af[i], bfr, acc[i][j]);
    }
  }

  for (int i = 0; i < 2; i++)
    for (int j = 0; j < 4; j++) {
      int n     = n0 + wn * 64 + j * 16 + l16;
      int mbase = m0 + wm * 32 + i * 16 + hf * 8;  // C layout rows (+8 hi half)
      float bv  = bias[n];
      if (MODE == 1) {
        for (int r = 0; r < 8; r++)
          outF[(size_t)(mbase + r) * Nd + n] = acc[i][j][r] + bv;
      } else {
        int which = n >> 10, rem = n & 1023, h = rem >> 6, d = rem & 63;
        uint16_t* dst = (which == 0) ? q_ws : ((which == 1) ? k_ws : v_ws);
        for (int r = 0; r < 8; r++) {
          int m = mbase + r;
          int b = m >> 11, t = m & 2047;
          int bh = b * NH + h;
          float v = acc[i][j][r] + bv;
          size_t idx = (which == 2) ? ((size_t)(bh * HD + d) * TT + t)   // V: [bh][d][t]
                                    : ((size_t)(bh * TT + t) * HD + d); // Q,K: [bh][t][d]
          dst[idx] = f2bf(v);
        }
      }
    }
}

// ---------------- flash attention (1 wave = 16 queries, full hd=64) ----------------

__global__ __launch_bounds__(128)
void flash_attn_wmma(const uint16_t* __restrict__ q_ws,
                     const uint16_t* __restrict__ k_ws,
                     const uint16_t* __restrict__ v_ws,
                     uint16_t* __restrict__ y_ws) {
  __shared__ uint16_t Plds[4][16 * 56];     // per-wave P staging (C->A transpose)

  const int tid  = threadIdx.x;
  const int lane = tid & 31;
  const int wave = tid >> 5;
  const int hf   = lane >> 4;
  const int l16  = lane & 15;
  const int bh   = blockIdx.x >> 5;                        // 64 (b,h) pairs
  // force q0 scalar so causal branches are SALU and EXEC stays all-1s (WMMA req.)
  const int q0   = __builtin_amdgcn_readfirstlane(((blockIdx.x & 31) * 4 + wave) * 16);
  const float scale = 0.125f;                              // 1/sqrt(64)

  // Q A-fragments, d = 0..31 and 32..63
  const uint16_t* qrow = q_ws + ((size_t)bh * TT + q0 + l16) * HD;
  v16bf qa0 = ldfragA(qrow + hf * 16);
  v16bf qa1 = ldfragA(qrow + 32 + hf * 16);

  float mrow[8], lrow[8], alpha[8];
  v8f acc[4];
  for (int r = 0; r < 8; r++) { mrow[r] = -1e30f; lrow[r] = 0.f; }
  for (int dt = 0; dt < 4; dt++) acc[dt] = zero8();

  const int ntiles = (q0 + 47) >> 5;   // ceil((q0+16)/32) 32-key tiles
  for (int kt = 0; kt < ntiles; kt++) {
    const int k0 = kt * 32;
    const uint16_t* krow = k_ws + ((size_t)bh * TT + k0 + l16) * HD;
    if (kt + 1 < ntiles)
      __builtin_prefetch(krow + 32 * HD, 0, 0);   // next K tile

    // ---- S = Q @ K^T  (two 16-key subtiles) ----
    v8f s0 = zero8();
    s0 = wmma_bf16(qa0, ldfragB(krow + hf * 16), s0);
    s0 = wmma_bf16(qa1, ldfragB(krow + 32 + hf * 16), s0);

    const bool has1 = (k0 + 16) <= (q0 + 15);     // scalar
    v8f s1 = zero8();
    if (has1) {
      const uint16_t* kr1 = krow + (size_t)16 * HD;
      s1 = wmma_bf16(qa0, ldfragB(kr1 + hf * 16), s1);
      s1 = wmma_bf16(qa1, ldfragB(kr1 + 32 + hf * 16), s1);
    }

    // scale + causal mask (uniform outer conditions)
    const bool mask0 = (k0 + 15) > q0;
    const bool mask1 = (k0 + 31) > q0;
    float S0[8], S1[8];
    for (int r = 0; r < 8; r++) {
      int row = q0 + r + hf * 8;
      float v0 = s0[r] * scale;
      if (mask0 && (k0 + l16) > row) v0 = -1e30f;
      S0[r] = v0;
      if (has1) {
        float v1 = s1[r] * scale;
        if (mask1 && (k0 + 16 + l16) > row) v1 = -1e30f;
        S1[r] = v1;
      } else {
        S1[r] = -1e30f;
      }
    }

    // ---- online softmax ----
    for (int r = 0; r < 8; r++) {
      float t = fmaxf(S0[r], S1[r]);
      t = fmaxf(t, __shfl_xor(t, 1));
      t = fmaxf(t, __shfl_xor(t, 2));
      t = fmaxf(t, __shfl_xor(t, 4));
      t = fmaxf(t, __shfl_xor(t, 8));      // row max over 16 lanes (stays in half)
      float mn = fmaxf(mrow[r], t);
      float al = __expf(mrow[r] - mn);
      mrow[r] = mn;
      alpha[r] = al;
      float p0 = __expf(S0[r] - mn);
      float p1 = __expf(S1[r] - mn);
      float sm = p0 + p1;
      sm += __shfl_xor(sm, 1);
      sm += __shfl_xor(sm, 2);
      sm += __shfl_xor(sm, 4);
      sm += __shfl_xor(sm, 8);
      lrow[r] = lrow[r] * al + sm;
      // transpose P into A-layout via LDS (per-wave DS ops are in-order)
      uint16_t* pp = Plds[wave] + (r + hf * 8) * 56;
      pp[l16]      = f2bf(p0);
      pp[16 + l16] = f2bf(p1);
    }
    for (int dt = 0; dt < 4; dt++)
      for (int r = 0; r < 8; r++) acc[dt][r] *= alpha[r];

    // ---- O += P @ V  (V stored [bh][d][t] -> contiguous B-fragments) ----
    v16bf pa = ldfragA(Plds[wave] + l16 * 56 + hf * 16);
    for (int dt = 0; dt < 4; dt++) {
      const uint16_t* vr =
          v_ws + ((size_t)(bh * HD + dt * 16 + l16)) * TT + k0 + hf * 16;
      acc[dt] = wmma_bf16(pa, ldfragB(vr), acc[dt]);
    }
  }

  // epilogue: normalize and write y (bf16, [B*T][C] layout for proj GEMM)
  const int b = bh >> 4, h = bh & 15;
  for (int r = 0; r < 8; r++) {
    float inv = 1.0f / lrow[r];
    int row = q0 + r + hf * 8;
    size_t ro = ((size_t)(b * TT + row)) * N_EMBD + h * HD;
    for (int dt = 0; dt < 4; dt++)
      y_ws[ro + dt * 16 + l16] = f2bf(acc[dt][r] * inv);
  }
}

// ---------------- launch ----------------

extern "C" void kernel_launch(void* const* d_in, const int* in_sizes, int n_in,
                              void* d_out, int out_size, void* d_ws, size_t ws_size,
                              hipStream_t stream) {
  (void)in_sizes; (void)n_in; (void)out_size; (void)ws_size;
  const float* x      = (const float*)d_in[0];
  const float* w_attn = (const float*)d_in[1];
  const float* b_attn = (const float*)d_in[2];
  const float* w_proj = (const float*)d_in[3];
  const float* b_proj = (const float*)d_in[4];
  float* out = (float*)d_out;

  char* ws = (char*)d_ws;
  size_t off = 0;
  uint16_t* xb  = (uint16_t*)(ws + off); off += (size_t)MM * N_EMBD * 2;         // x bf16
  uint16_t* wat = (uint16_t*)(ws + off); off += (size_t)3 * N_EMBD * N_EMBD * 2; // w_attn^T
  uint16_t* wpt = (uint16_t*)(ws + off); off += (size_t)N_EMBD * N_EMBD * 2;     // w_proj^T
  uint16_t* qw  = (uint16_t*)(ws + off); off += (size_t)MM * N_EMBD * 2;         // Q
  uint16_t* kw  = (uint16_t*)(ws + off); off += (size_t)MM * N_EMBD * 2;         // K
  uint16_t* vw  = (uint16_t*)(ws + off); off += (size_t)MM * N_EMBD * 2;         // V^T
  uint16_t* yb  = (uint16_t*)(ws + off); off += (size_t)MM * N_EMBD * 2;         // attn out

  // prep: precision downcast + weight transposes (one-time, bandwidth-trivial)
  cvt_f32_bf16<<<(MM * N_EMBD) / 1024, 256, 0, stream>>>(x, xb, MM * N_EMBD);
  transpose_cvt<<<dim3(3 * N_EMBD / 32, N_EMBD / 32), 256, 0, stream>>>(
      w_attn, wat, N_EMBD, 3 * N_EMBD);
  transpose_cvt<<<dim3(N_EMBD / 32, N_EMBD / 32), 256, 0, stream>>>(
      w_proj, wpt, N_EMBD, N_EMBD);

  // QKV projection: [8192,1024] x [1024,3072] -> per-head Q,K,V (V transposed)
  gemm_bf16_wmma<0><<<dim3(3 * N_EMBD / 128, MM / 128), 256, 0, stream>>>(
      xb, wat, b_attn, nullptr, qw, kw, vw, 3 * N_EMBD, N_EMBD);

  // flash attention: 64 (b,h) x 128 q-tiles, 4 waves/block
  flash_attn_wmma<<<dim3(64 * 32), 128, 0, stream>>>(qw, kw, vw, yb);

  // output projection: [8192,1024] x [1024,1024] -> f32 out
  gemm_bf16_wmma<1><<<dim3(N_EMBD / 128, MM / 128), 256, 0, stream>>>(
      yb, wpt, b_proj, out, nullptr, nullptr, nullptr, N_EMBD, N_EMBD);
}
